// MultiheadAttention_25486335935050
// MI455X (gfx1250) — matrix operand order repaired and checked by reading, not compiled, not run on hardware
//
#include <hip/hip_runtime.h>
#include <hip/hip_bf16.h>

typedef __bf16 bf16_t;
typedef __attribute__((ext_vector_type(16))) __bf16 v16bf;
typedef __attribute__((ext_vector_type(8)))  __bf16 v8bf;
typedef __attribute__((ext_vector_type(4)))  __bf16 v4bf;
typedef __attribute__((ext_vector_type(8)))  float  v8f;
typedef __attribute__((ext_vector_type(4)))  float  v4f;

// ---------------------------------------------------------------------------
// fp32 -> bf16 conversion, vectorized: float4 in -> 4x bf16 (b64) out.
// n must be a multiple of 4 (all tensors here are multiples of 1024).
// ---------------------------------------------------------------------------
__global__ __launch_bounds__(256) void cvt_f32_bf16(const float* __restrict__ src,
                                                    bf16_t* __restrict__ dst, int n4) {
  int i = blockIdx.x * blockDim.x + threadIdx.x;
  const int stride = gridDim.x * blockDim.x;
  const v4f* s4 = (const v4f*)src;
  v4bf* d4 = (v4bf*)dst;
  for (; i < n4; i += stride) {
    v4f x = s4[i];
    v4bf y;
#pragma unroll
    for (int j = 0; j < 4; j++) y[j] = (bf16_t)x[j];
    d4[i] = y;
  }
}

// ---------------------------------------------------------------------------
// C[M,N] = A[M,K] @ W[N,K]^T + bias   (torch nn.Linear), bf16 in / f32 out.
// Block tile 128x128, 8 waves (2 M x 4 N), each wave: 64x32 = 4x2 WMMA frags.
// Software-pipelined: fragments for step k+1 are loaded before the WMMAs of
// step k are issued (register double buffer), so s_wait_loadcnt covers loads
// issued a full MMA step earlier.
//
// v_wmma_f32_16x16x32_bf16 VGPR layouts (CDNA5 ISA 7.12.2):
//   A 16x32: lane L -> row L%16, K-halves {8*(L/16) .. +7} and {+16 .. +23}
//   B 32x16: lane L -> col L%16, K-halves {8*(L/16) .. +7} and {+16 .. +23} (same as A)
//   C 16x16: lane L, vgpr r -> row 8*(L/16)+r, col L%16
// ---------------------------------------------------------------------------
__device__ __forceinline__ v16bf load_a_frag(const bf16_t* p) {
  v8bf lo = *(const v8bf*)p;         // K = kbase    .. +7
  v8bf hi = *(const v8bf*)(p + 16);  // K = kbase+16 .. +23
  return __builtin_shufflevector(lo, hi, 0, 1, 2, 3, 4, 5, 6, 7,
                                 8, 9, 10, 11, 12, 13, 14, 15);
}

__global__ __launch_bounds__(256) void gemm_qkv(const bf16_t* __restrict__ A,
                                                const bf16_t* __restrict__ W,
                                                const float* __restrict__ bias,
                                                float* __restrict__ C,
                                                int M, int N, int K) {
  const int lane  = threadIdx.x & 31;
  const int wid   = threadIdx.x >> 5;
  const int waveM = wid >> 2;   // 0..1
  const int waveN = wid & 3;    // 0..3
  const int l16   = lane & 15;
  const int lhalf = lane >> 4;  // 0/1

  const int rowBase = blockIdx.y * 128 + waveM * 64;
  const int colBase = blockIdx.x * 128 + waveN * 32;

  const int akb = lhalf * 8;    // A K sub-offset for this lane
  const int bkb = lhalf * 8;    // B K sub-offset for this lane

  v8f acc[4][2] = {};

  const bf16_t* aptr[4];
#pragma unroll
  for (int m = 0; m < 4; m++)
    aptr[m] = A + (size_t)(rowBase + m * 16 + l16) * K + akb;
  const bf16_t* bptr[2];
#pragma unroll
  for (int n = 0; n < 2; n++)
    bptr[n] = W + (size_t)(colBase + n * 16 + l16) * K + bkb;

#define LOAD_STEP(af, bf, kk)                                              \
  do {                                                                     \
    _Pragma("unroll") for (int m = 0; m < 4; m++)                          \
        af[m] = load_a_frag(aptr[m] + (kk));                               \
    _Pragma("unroll") for (int n = 0; n < 2; n++)                          \
        bf[n] = load_a_frag(bptr[n] + (kk));                               \
  } while (0)

#define MMA_STEP(af, bf)                                                   \
  do {                                                                     \
    _Pragma("unroll") for (int m = 0; m < 4; m++)                          \
        _Pragma("unroll") for (int n = 0; n < 2; n++)                      \
            acc[m][n] = __builtin_amdgcn_wmma_f32_16x16x32_bf16(           \
                false, af[m], false, bf[n], (short)0, acc[m][n],           \
                false, false);                                             \
  } while (0)

  v16bf a0[4], b0[2], a1[4], b1[2];
  const int nsteps = K >> 5;  // K/32, even for K=1024

  LOAD_STEP(a0, b0, 0);
#pragma unroll 1
  for (int s = 0; s < nsteps - 2; s += 2) {
    LOAD_STEP(a1, b1, (s + 1) * 32);
    MMA_STEP(a0, b0);
    LOAD_STEP(a0, b0, (s + 2) * 32);
    MMA_STEP(a1, b1);
  }
  LOAD_STEP(a1, b1, (nsteps - 1) * 32);
  MMA_STEP(a0, b0);
  MMA_STEP(a1, b1);

#undef LOAD_STEP
#undef MMA_STEP

  // Epilogue: bias add + f32 store per C/D layout (lanes 0-15 cover 16
  // contiguous columns -> coalesced 64B per row-store across the half-wave).
#pragma unroll
  for (int n = 0; n < 2; n++) {
    const int col = colBase + n * 16 + l16;
    const float bv = bias[col];
#pragma unroll
    for (int m = 0; m < 4; m++) {
      const int row0 = rowBase + m * 16 + lhalf * 8;
      float* cp = C + (size_t)row0 * N + col;
#pragma unroll
      for (int r = 0; r < 8; r++) cp[(size_t)r * N] = acc[m][n][r] + bv;
    }
  }
}

// ---------------------------------------------------------------------------
// Fused relu-outer-product attention + head-interleave concat + residual + LN.
// relu(qp_i*kp_j/sqrt(hd))*vp_j summed over j collapses to
//   qp_i/sqrt(hd) * (qp_i>0 ? sum_{kp_j>0} kp_j*vp_j : sum_{kp_j<0} kp_j*vp_j)
// One 256-thread block per batch row. nhid=1024, H=32, hd=32 assumed.
// ---------------------------------------------------------------------------
__global__ __launch_bounds__(256) void attn_ln(const float* __restrict__ qp,
                                               const float* __restrict__ kp,
                                               const float* __restrict__ vp,
                                               const float* __restrict__ qres,
                                               const float* __restrict__ gamma,
                                               const float* __restrict__ beta,
                                               float* __restrict__ out, int nhid) {
  const int b = blockIdx.x;
  const int t = threadIdx.x;
  __shared__ float s_pos[32], s_neg[32];
  __shared__ float rs[256], rq[256];

  const float* kprow = kp + (size_t)b * nhid;
  const float* vprow = vp + (size_t)b * nhid;
  const float* qprow = qp + (size_t)b * nhid;
  const float* qrow  = qres + (size_t)b * nhid;

  // per-head sign-partitioned sums of kp_j * vp_j  (8 threads per head)
  const int h = t >> 3, sub = t & 7;
  float pos = 0.f, neg = 0.f;
#pragma unroll
  for (int jj = 0; jj < 4; jj++) {
    const int j = sub * 4 + jj;
    const float kv = kprow[h * 32 + j];
    const float vv = vprow[h * 32 + j];
    const float p = kv * vv;
    if (kv > 0.f) pos += p; else neg += p;
  }
#pragma unroll
  for (int m = 1; m < 8; m <<= 1) {  // 8 consecutive lanes, same wave32
    pos += __shfl_xor(pos, m, 32);
    neg += __shfl_xor(neg, m, 32);
  }
  if (sub == 0) { s_pos[h] = pos; s_neg[h] = neg; }
  __syncthreads();

  const float scale = 0.17677669529663687f;  // 1/sqrt(32)
  float x[4];
  float lsum = 0.f, lsq = 0.f;
#pragma unroll
  for (int ii = 0; ii < 4; ii++) {
    const int f = t * 4 + ii;  // concat feature index
    const int hh = f & 31;     // f = i*H + h  (transpose(1,2) interleave)
    const int i = f >> 5;
    const float qv = qprow[hh * 32 + i];
    const float S = (qv > 0.f) ? s_pos[hh] : s_neg[hh];
    const float xv = qv * scale * S + qrow[f];
    x[ii] = xv;
    lsum += xv;
    lsq += xv * xv;
  }
  rs[t] = lsum; rq[t] = lsq;
  __syncthreads();
  for (int off = 128; off > 0; off >>= 1) {
    if (t < off) { rs[t] += rs[t + off]; rq[t] += rq[t + off]; }
    __syncthreads();
  }
  const float mean = rs[0] / (float)nhid;
  const float var  = rq[0] / (float)nhid - mean * mean;
  const float inv  = rsqrtf(var + 1e-5f);
#pragma unroll
  for (int ii = 0; ii < 4; ii++) {
    const int f = t * 4 + ii;
    out[(size_t)b * nhid + f] = (x[ii] - mean) * inv * gamma[f] + beta[f];
  }
}

// ---------------------------------------------------------------------------
extern "C" void kernel_launch(void* const* d_in, const int* in_sizes, int n_in,
                              void* d_out, int out_size, void* d_ws, size_t ws_size,
                              hipStream_t stream) {
  const float* q  = (const float*)d_in[0];
  const float* k  = (const float*)d_in[1];
  const float* v  = (const float*)d_in[2];
  const float* Wq = (const float*)d_in[3];
  const float* bq = (const float*)d_in[4];
  const float* Wk = (const float*)d_in[5];
  const float* bk = (const float*)d_in[6];
  const float* Wv = (const float*)d_in[7];
  const float* bv = (const float*)d_in[8];
  const float* gamma = (const float*)d_in[9];
  const float* beta  = (const float*)d_in[10];

  const int nhid = in_sizes[4];      // 1024
  const int M = in_sizes[0] / nhid;  // 8192
  const int N = nhid, K = nhid;

  // Workspace carve-up
  char* ws = (char*)d_ws;
  size_t off = 0;
  auto carve = [&](size_t bytes) -> void* {
    void* p = ws + off;
    off = (off + bytes + 255) & ~(size_t)255;
    return p;
  };
  bf16_t* qb  = (bf16_t*)carve((size_t)M * K * sizeof(bf16_t));
  bf16_t* kb  = (bf16_t*)carve((size_t)M * K * sizeof(bf16_t));
  bf16_t* vb  = (bf16_t*)carve((size_t)M * K * sizeof(bf16_t));
  bf16_t* Wqb = (bf16_t*)carve((size_t)N * K * sizeof(bf16_t));
  bf16_t* Wkb = (bf16_t*)carve((size_t)N * K * sizeof(bf16_t));
  bf16_t* Wvb = (bf16_t*)carve((size_t)N * K * sizeof(bf16_t));
  float*  qpj = (float*)carve((size_t)M * N * sizeof(float));
  float*  kpj = (float*)carve((size_t)M * N * sizeof(float));
  float*  vpj = (float*)carve((size_t)M * N * sizeof(float));

  const int nMK4 = (M * K) / 4;
  const int nNK4 = (N * K) / 4;
  const int cvtBlocks = 2048;  // grid-stride

  cvt_f32_bf16<<<cvtBlocks, 256, 0, stream>>>(q,  qb,  nMK4);
  cvt_f32_bf16<<<cvtBlocks, 256, 0, stream>>>(k,  kb,  nMK4);
  cvt_f32_bf16<<<cvtBlocks, 256, 0, stream>>>(v,  vb,  nMK4);
  cvt_f32_bf16<<<cvtBlocks, 256, 0, stream>>>(Wq, Wqb, nNK4);
  cvt_f32_bf16<<<cvtBlocks, 256, 0, stream>>>(Wk, Wkb, nNK4);
  cvt_f32_bf16<<<cvtBlocks, 256, 0, stream>>>(Wv, Wvb, nNK4);

  dim3 gGrid(N / 128, M / 128);
  gemm_qkv<<<gGrid, 256, 0, stream>>>(qb, Wqb, bq, qpj, M, N, K);
  gemm_qkv<<<gGrid, 256, 0, stream>>>(kb, Wkb, bk, kpj, M, N, K);
  gemm_qkv<<<gGrid, 256, 0, stream>>>(vb, Wvb, bv, vpj, M, N, K);

  attn_ln<<<M, 256, 0, stream>>>(qpj, kpj, vpj, q, gamma, beta, (float*)d_out, nhid);
}
